// LongShortTermTransformer_33354716021309
// MI455X (gfx1250) — matrix operand order repaired and checked by reading, not compiled, not run on hardware
//
#include <hip/hip_runtime.h>

#define N_B    4
#define C_ALL  256
#define N_HEAD 2
#define HDIM   128
#define MAXD   7
#define WSIDE  15
#define WWIN   225
#define HW     1024
#define PH     46            // padded rows: 32 + 2*7
#define PWX    48            // padded x-stride: 46 rounded to 48 for b128 loads
#define PADA   (PH * PWX)
#define REL_STR 228          // rel w-stride (f32, 16B aligned rows)
#define ATT_STR 240          // attn w-stride (bf16, 16B aligned rows)

typedef __bf16 bf16_t;
typedef __attribute__((ext_vector_type(16))) __bf16 v16bf;
typedef __attribute__((ext_vector_type(8)))  __bf16 v8bf;
typedef __attribute__((ext_vector_type(8)))  float  v8f;
typedef __attribute__((ext_vector_type(4)))  float  v4f;

__device__ __forceinline__ v8f wmma_bf16(const v16bf& a, const v16bf& b, const v8f& c) {
  return __builtin_amdgcn_wmma_f32_16x16x32_bf16(false, a, false, b, (short)0, c, false, false);
}

// A-fragment: row-major [M][K] tile, K contiguous. Lane reads 2 runs of 8.
__device__ __forceinline__ v16bf load_frag_a(const bf16_t* rowp, int half) {
  union { v16bf v; v8bf h[2]; } u;
  u.h[0] = *(const v8bf*)(rowp + half * 8);
  u.h[1] = *(const v8bf*)(rowp + 16 + half * 8);
  return u.v;
}
// B-fragment: transposed [N][K] tile, K contiguous. Lane reads 2 runs of 8.
__device__ __forceinline__ v16bf load_frag_b(const bf16_t* colp, int half) {
  union { v16bf v; v8bf h[2]; } u;
  u.h[0] = *(const v8bf*)(colp + half * 8);
  u.h[1] = *(const v8bf*)(colp + 16 + half * 8);
  return u.v;
}
__device__ __forceinline__ v8bf cvt8_from(const float* g) {
  v4f a = *(const v4f*)g;
  v4f b = *(const v4f*)(g + 4);
  v8bf r;
  for (int i = 0; i < 4; ++i) { r[i] = (bf16_t)a[i]; r[4 + i] = (bf16_t)b[i]; }
  return r;
}

// ---------------------------------------------------------------------------
// Projection GEMM: Y[d,m] = sum_c W[d,c] X[n,c,m] + b[d]
// Block tile 128x128, wave tile 32x64 (2M x 4N), K-step 32.
// mode 0: Qt[n][m][c]   mode 1: KpadT[n][g][yy][xx][d]   mode 2: Vpad[n][c][yy*48+xx]
// ---------------------------------------------------------------------------
__global__ __launch_bounds__(256)
void proj_gemm_kernel(const float* __restrict__ X, const float* __restrict__ Wm,
                      const float* __restrict__ bias, bf16_t* __restrict__ Y, int mode) {
  __shared__ bf16_t aW[128][40];
  __shared__ bf16_t bXT[128][40];
  const int tid = threadIdx.x;
  const int wave = tid >> 5, lane = tid & 31, half = lane >> 4, l16 = lane & 15;
  const int mq = wave & 3, nh = wave >> 2;
  const int mbase = blockIdx.x * 128;
  const int nbase = blockIdx.y * 128;
  const int n = blockIdx.z;
  v8f acc[2][4];
  for (int i = 0; i < 2; ++i)
    for (int t = 0; t < 4; ++t) acc[i][t] = (v8f){0.f,0.f,0.f,0.f,0.f,0.f,0.f,0.f};

  for (int k0 = 0; k0 < C_ALL; k0 += 32) {
    {   // A: 128x32, thread -> 16 contiguous
      int r = tid >> 1, c0 = (tid & 1) * 16;
      const float* src = &Wm[(size_t)(mbase + r) * C_ALL + k0 + c0];
      *(v8bf*)&aW[r][c0]     = cvt8_from(src);
      *(v8bf*)&aW[r][c0 + 8] = cvt8_from(src + 8);
    }
    for (int t = 0; t < 2; ++t) {   // B transposed: 128 cols x 32 k, coalesced gather
      int e = tid + 256 * t;
      int m = e & 127, kk0 = (e >> 7) * 8;
      v8bf tmp;
      for (int i = 0; i < 8; ++i)
        tmp[i] = (bf16_t)X[(size_t)(n * C_ALL + k0 + kk0 + i) * HW + nbase + m];
      *(v8bf*)&bXT[m][kk0] = tmp;
    }
    __syncthreads();
    v16bf a0 = load_frag_a(&aW[mq * 32 + l16][0], half);
    v16bf a1 = load_frag_a(&aW[mq * 32 + 16 + l16][0], half);
    for (int nt = 0; nt < 4; ++nt) {
      v16bf b = load_frag_b(&bXT[nh * 64 + nt * 16 + l16][0], half);
      acc[0][nt] = wmma_bf16(a0, b, acc[0][nt]);
      acc[1][nt] = wmma_bf16(a1, b, acc[1][nt]);
    }
    __syncthreads();
  }
  for (int mi = 0; mi < 2; ++mi) {
    const int rowbase = mbase + mq * 32 + mi * 16 + 8 * half;   // 8 contiguous rows
    v4f bi0 = *(const v4f*)&bias[rowbase];
    v4f bi1 = *(const v4f*)&bias[rowbase + 4];
    for (int nt = 0; nt < 4; ++nt) {
      int col = nbase + nh * 64 + nt * 16 + l16;
      float vv[8];
      for (int r = 0; r < 4; ++r) {
        vv[r] = acc[mi][nt][r] + bi0[r];
        vv[4 + r] = acc[mi][nt][4 + r] + bi1[r];
      }
      if (mode == 0) {
        v8bf o; for (int r = 0; r < 8; ++r) o[r] = (bf16_t)vv[r];
        *(v8bf*)&Y[((size_t)n * HW + col) * C_ALL + rowbase] = o;
      } else if (mode == 1) {
        int yy = col >> 5, xx = col & 31;
        int g = rowbase >> 7, d0 = rowbase & 127;
        v8bf o; for (int r = 0; r < 8; ++r) o[r] = (bf16_t)vv[r];
        size_t idx = ((((size_t)(n * 2 + g) * PH) + yy + MAXD) * PWX + xx + MAXD) * HDIM + d0;
        *(v8bf*)&Y[idx] = o;
      } else {
        int yy = col >> 5, xx = col & 31;
        for (int r = 0; r < 8; ++r)
          Y[(size_t)(n * C_ALL + rowbase + r) * PADA + (yy + MAXD) * PWX + (xx + MAXD)] = (bf16_t)vv[r];
      }
    }
  }
}

// ---------------------------------------------------------------------------
// rel[n,g,w,m] = sum_d Wrel[g,w,d] * q[n,g,d,m] + brel  -> relT[z][m][w]
// ---------------------------------------------------------------------------
__global__ __launch_bounds__(256)
void rel_gemm_kernel(const bf16_t* __restrict__ Qt, const float* __restrict__ Wrel,
                     const float* __restrict__ brel, float* __restrict__ relT) {
  __shared__ bf16_t aW[128][40];
  __shared__ bf16_t bXT[128][40];
  const int tid = threadIdx.x;
  const int wave = tid >> 5, lane = tid & 31, half = lane >> 4, l16 = lane & 15;
  const int mq = wave & 3, nh = wave >> 2;
  const int mbase = blockIdx.x * 128;
  const int nbase = blockIdx.y * 128;
  const int z = blockIdx.z;
  const int n = z >> 1, g = z & 1;
  v8f acc[2][4];
  for (int i = 0; i < 2; ++i)
    for (int t = 0; t < 4; ++t) acc[i][t] = (v8f){0.f,0.f,0.f,0.f,0.f,0.f,0.f,0.f};

  for (int k0 = 0; k0 < HDIM; k0 += 32) {
    {
      int r = tid >> 1, c0 = (tid & 1) * 16;
      int w = mbase + r;
      if (w < WWIN) {
        const float* src = &Wrel[((size_t)g * WWIN + w) * HDIM + k0 + c0];
        *(v8bf*)&aW[r][c0]     = cvt8_from(src);
        *(v8bf*)&aW[r][c0 + 8] = cvt8_from(src + 8);
      } else {
        v8bf zz = (v8bf)(bf16_t)0.f;
        *(v8bf*)&aW[r][c0] = zz;
        *(v8bf*)&aW[r][c0 + 8] = zz;
      }
    }
    for (int t = 0; t < 2; ++t) {
      int e = tid + 256 * t;
      int m = e >> 2, c8 = (e & 3) * 8;
      *(v8bf*)&bXT[m][c8] =
        *(const v8bf*)&Qt[((size_t)n * HW + nbase + m) * C_ALL + g * HDIM + k0 + c8];
    }
    __syncthreads();
    v16bf a0 = load_frag_a(&aW[mq * 32 + l16][0], half);
    v16bf a1 = load_frag_a(&aW[mq * 32 + 16 + l16][0], half);
    for (int nt = 0; nt < 4; ++nt) {
      v16bf b = load_frag_b(&bXT[nh * 64 + nt * 16 + l16][0], half);
      acc[0][nt] = wmma_bf16(a0, b, acc[0][nt]);
      acc[1][nt] = wmma_bf16(a1, b, acc[1][nt]);
    }
    __syncthreads();
  }
  for (int mi = 0; mi < 2; ++mi) {
    const int rowbase = mbase + mq * 32 + mi * 16 + 8 * half;
    for (int nt = 0; nt < 4; ++nt) {
      int col = nbase + nh * 64 + nt * 16 + l16;
      size_t obase = ((size_t)z * HW + col) * REL_STR;
      for (int r = 0; r < 8; ++r) {
        int row = rowbase + r;
        if (row < WWIN) relT[obase + row] = acc[mi][nt][r] + brel[g * WWIN + row];
      }
    }
  }
}

// ---------------------------------------------------------------------------
// qk via banded Gram + mask + rel + softmax -> attnT[z][m][w]
// ---------------------------------------------------------------------------
__global__ __launch_bounds__(64)
void qk_softmax_kernel(const bf16_t* __restrict__ Qt, const bf16_t* __restrict__ KpadT,
                       const float* __restrict__ relT, bf16_t* __restrict__ attnT) {
  __shared__ bf16_t q_ldsT[16][136];
  __shared__ bf16_t k_ldsT[32][136];
  __shared__ float  gram[16][WWIN + 1];
  const int tid = threadIdx.x;
  const int hwv = tid >> 5, lane = tid & 31, half = lane >> 4, l16 = lane & 15;
  const int x0 = blockIdx.x * 16;
  const int y  = blockIdx.y;
  const int z  = blockIdx.z;
  const int n = z >> 1, g = z & 1;

  for (int t = 0; t < 4; ++t) {               // Q tile: 16 x 128 b128 chunks
    int e = tid + 64 * t;
    int xq = e >> 4, c8 = (e & 15) * 8;
    *(v8bf*)&q_ldsT[xq][c8] =
      *(const v8bf*)&Qt[((size_t)n * HW + y * 32 + x0 + xq) * C_ALL + g * HDIM + c8];
  }
  const size_t krow = (size_t)(n * 2 + g) * PH;
  for (int dy = 0; dy < WSIDE; ++dy) {
    __syncthreads();
    for (int t = 0; t < 8; ++t) {             // Ksub rows j=0..31, d contiguous
      int e = tid + 64 * t;
      int j = e >> 4, c8 = (e & 15) * 8;
      *(v8bf*)&k_ldsT[j][c8] =
        *(const v8bf*)&KpadT[((krow + y + dy) * PWX + x0 + j) * HDIM + c8];
    }
    __syncthreads();
    v8f acc = (v8f){0.f,0.f,0.f,0.f,0.f,0.f,0.f,0.f};
    for (int ks = 0; ks < 4; ++ks) {
      v16bf a = load_frag_a(&k_ldsT[hwv * 16 + l16][ks * 32], half);
      v16bf b = load_frag_b(&q_ldsT[l16][ks * 32], half);
      acc = wmma_bf16(a, b, acc);
    }
    for (int r = 0; r < 8; ++r) {             // extract diagonal band
      int j = hwv * 16 + r + 8 * half;
      int dx = j - l16;
      if ((unsigned)dx < 15u) gram[l16][dy * 15 + dx] = acc[r];
    }
  }
  __syncthreads();
  if (tid < 16) {
    const int p = tid;
    const int m = y * 32 + x0 + p;
    const float invT = 0.08838834764831845f;  // 1/sqrt(128)
    const size_t rbase = ((size_t)z * HW + m) * REL_STR;
    float mx = -3.0e38f;
    for (int w = 0; w < WWIN; ++w) {
      int dy = w / 15, dx = w % 15;
      int ky = y + dy - MAXD, kx = x0 + p + dx - MAXD;
      float vv = ((unsigned)ky < 32u && (unsigned)kx < 32u)
               ? gram[p][w] * invT + relT[rbase + w]
               : -1.0e8f;
      gram[p][w] = vv;
      mx = fmaxf(mx, vv);
    }
    float s = 0.f;
    for (int w = 0; w < WWIN; ++w) { float e = __expf(gram[p][w] - mx); gram[p][w] = e; s += e; }
    float rs = 1.f / s;
    const size_t abase = ((size_t)z * HW + m) * ATT_STR;
    for (int w = 0; w < WWIN; ++w) attnT[abase + w] = (bf16_t)(gram[p][w] * rs);
  }
}

// ---------------------------------------------------------------------------
// agg = V_bias x attn + unfolded(V) x banded(attn)  -> aggT[col][c]
// ---------------------------------------------------------------------------
__global__ __launch_bounds__(256)
void agg_kernel(const bf16_t* __restrict__ attnT, const bf16_t* __restrict__ Vpad,
                const bf16_t* __restrict__ Vbbf, bf16_t* __restrict__ aggT) {
  __shared__ bf16_t at_ldsT[16][264];
  __shared__ bf16_t v_lds[128][40];
  __shared__ bf16_t s_ldsT[16][40];
  const int tid = threadIdx.x;
  const int wave = tid >> 5, lane = tid & 31, half = lane >> 4, l16 = lane & 15;
  const int x0 = blockIdx.x * 16;
  const int y  = blockIdx.y;
  const int z  = blockIdx.z;
  const int n = z >> 1, g = z & 1;
  const int m0 = y * 32 + x0;

  for (int t = 0; t < 2; ++t) {               // attn tile [16][256], w contiguous
    int e = tid + 256 * t;
    int xq = e >> 5, ci = e & 31;
    v8bf val;
    if (ci < 30) val = *(const v8bf*)&attnT[((size_t)z * HW + m0 + xq) * ATT_STR + ci * 8];
    else val = (v8bf)(bf16_t)0.f;             // pad w=240..255 (Vbbf cols are zero >=225)
    *(v8bf*)&at_ldsT[xq][ci * 8] = val;
  }
  v8f acc = (v8f){0.f,0.f,0.f,0.f,0.f,0.f,0.f,0.f};

  // Phase 1: agg_bias = V_bias(128x256pad) x attn(256x16)
  for (int ks = 0; ks < 8; ++ks) {
    __syncthreads();
    for (int t = 0; t < 2; ++t) {
      int e = tid + 256 * t;
      int d = e >> 2, c8 = (e & 3) * 8;
      *(v8bf*)&v_lds[d][c8] = *(const v8bf*)&Vbbf[(size_t)(g * HDIM + d) * 256 + ks * 32 + c8];
    }
    __syncthreads();
    v16bf a = load_frag_a(&v_lds[wave * 16 + l16][0], half);
    v16bf b = load_frag_b(&at_ldsT[l16][ks * 32], half);
    acc = wmma_bf16(a, b, acc);
  }

  // Phase 2: agg_value, banded scatter of attn per dy
  const size_t vbase = (size_t)(n * C_ALL + g * HDIM) * PADA;
  for (int dy = 0; dy < WSIDE; ++dy) {
    __syncthreads();
    for (int t = 0; t < 2; ++t) {
      int e = tid + 256 * t;
      int d = e >> 2, j8 = (e & 3) * 8;
      *(v8bf*)&v_lds[d][j8] =
        *(const v8bf*)&Vpad[vbase + (size_t)d * PADA + (y + dy) * PWX + x0 + j8];
    }
    for (int t = 0; t < 2; ++t) {             // banded B operand [xq][j]
      int e = tid + 256 * t;
      int xq = e >> 5, j = e & 31;
      int dx = j - xq;
      s_ldsT[xq][j] = ((unsigned)dx < 15u) ? at_ldsT[xq][dy * 15 + dx] : (bf16_t)0.f;
    }
    __syncthreads();
    v16bf a = load_frag_a(&v_lds[wave * 16 + l16][0], half);
    v16bf b = load_frag_b(&s_ldsT[l16][0], half);
    acc = wmma_bf16(a, b, acc);
  }

  const int dbase = wave * 16 + 8 * half;     // 8 contiguous channels
  v8bf o; for (int r = 0; r < 8; ++r) o[r] = (bf16_t)acc[r];
  int col = n * HW + m0 + l16;
  *(v8bf*)&aggT[(size_t)col * C_ALL + g * HDIM + dbase] = o;
}

// ---------------------------------------------------------------------------
// Final fc: out[m,n,c'] = sum_c Wfc[c',c] * aggT[n*1024+m][c] + bfc[c']
// ---------------------------------------------------------------------------
__global__ __launch_bounds__(256)
void fc_gemm_kernel(const bf16_t* __restrict__ aggT, const float* __restrict__ Wfc,
                    const float* __restrict__ bfc, float* __restrict__ out) {
  __shared__ bf16_t aW[128][40];
  __shared__ bf16_t bXT[128][40];
  const int tid = threadIdx.x;
  const int wave = tid >> 5, lane = tid & 31, half = lane >> 4, l16 = lane & 15;
  const int mq = wave & 3, nh = wave >> 2;
  const int mbase = blockIdx.x * 128;
  const int nbase = blockIdx.y * 128;
  v8f acc[2][4];
  for (int i = 0; i < 2; ++i)
    for (int t = 0; t < 4; ++t) acc[i][t] = (v8f){0.f,0.f,0.f,0.f,0.f,0.f,0.f,0.f};

  for (int k0 = 0; k0 < C_ALL; k0 += 32) {
    {
      int r = tid >> 1, c0 = (tid & 1) * 16;
      const float* src = &Wfc[(size_t)(mbase + r) * C_ALL + k0 + c0];
      *(v8bf*)&aW[r][c0]     = cvt8_from(src);
      *(v8bf*)&aW[r][c0 + 8] = cvt8_from(src + 8);
    }
    for (int t = 0; t < 2; ++t) {
      int e = tid + 256 * t;
      int m = e >> 2, c8 = (e & 3) * 8;
      *(v8bf*)&bXT[m][c8] = *(const v8bf*)&aggT[(size_t)(nbase + m) * C_ALL + k0 + c8];
    }
    __syncthreads();
    v16bf a0 = load_frag_a(&aW[mq * 32 + l16][0], half);
    v16bf a1 = load_frag_a(&aW[mq * 32 + 16 + l16][0], half);
    for (int nt = 0; nt < 4; ++nt) {
      v16bf b = load_frag_b(&bXT[nh * 64 + nt * 16 + l16][0], half);
      acc[0][nt] = wmma_bf16(a0, b, acc[0][nt]);
      acc[1][nt] = wmma_bf16(a1, b, acc[1][nt]);
    }
    __syncthreads();
  }
  for (int mi = 0; mi < 2; ++mi) {
    const int rowbase = mbase + mq * 32 + mi * 16 + 8 * half;
    v4f bi0 = *(const v4f*)&bfc[rowbase];
    v4f bi1 = *(const v4f*)&bfc[rowbase + 4];
    for (int nt = 0; nt < 4; ++nt) {
      int col = nbase + nh * 64 + nt * 16 + l16;
      int nn = col >> 10, m = col & 1023;
      v4f o0, o1;
      for (int r = 0; r < 4; ++r) {
        o0[r] = acc[mi][nt][r] + bi0[r];
        o1[r] = acc[mi][nt][4 + r] + bi1[r];
      }
      size_t idx = ((size_t)m * N_B + nn) * C_ALL + rowbase;
      *(v4f*)&out[idx]     = o0;
      *(v4f*)&out[idx + 4] = o1;
    }
  }
}

__global__ __launch_bounds__(256)
void vb_convert_kernel(const float* __restrict__ Vb, bf16_t* __restrict__ outp) {
  int e = blockIdx.x * 256 + threadIdx.x;     // 2*128*256 = 65536
  int w = e & 255, gd = e >> 8;
  outp[e] = (w < WWIN) ? (bf16_t)Vb[(size_t)gd * WWIN + w] : (bf16_t)0.f;
}

// ---------------------------------------------------------------------------
extern "C" void kernel_launch(void* const* d_in, const int* in_sizes, int n_in,
                              void* d_out, int out_size, void* d_ws, size_t ws_size,
                              hipStream_t stream) {
  (void)in_sizes; (void)n_in; (void)out_size; (void)ws_size;
  const float* q    = (const float*)d_in[0];
  const float* k    = (const float*)d_in[1];
  const float* v    = (const float*)d_in[2];
  const float* Wq   = (const float*)d_in[3];
  const float* bq   = (const float*)d_in[4];
  const float* Wk   = (const float*)d_in[5];
  const float* bk   = (const float*)d_in[6];
  const float* Wv   = (const float*)d_in[7];
  const float* bv   = (const float*)d_in[8];
  const float* Wrel = (const float*)d_in[9];
  const float* brel = (const float*)d_in[10];
  const float* Vb   = (const float*)d_in[11];
  const float* Wfc  = (const float*)d_in[12];
  const float* bfc  = (const float*)d_in[13];
  float* out = (float*)d_out;

  char* ws = (char*)d_ws;
  size_t off = 0;
  auto carve = [&](size_t bytes) -> void* {
    void* p = ws + off;
    off = (off + bytes + 255) & ~((size_t)255);
    return p;
  };
  bf16_t* Qt    = (bf16_t*)carve((size_t)N_B * HW * C_ALL * 2);
  bf16_t* KpadT = (bf16_t*)carve((size_t)N_B * N_HEAD * PADA * HDIM * 2);
  bf16_t* Vpad  = (bf16_t*)carve((size_t)N_B * C_ALL * PADA * 2);
  float*  relT  = (float*) carve((size_t)N_B * N_HEAD * HW * REL_STR * 4);
  bf16_t* attnT = (bf16_t*)carve((size_t)N_B * N_HEAD * HW * ATT_STR * 2);
  bf16_t* Vbbf  = (bf16_t*)carve((size_t)N_HEAD * HDIM * 256 * 2);
  bf16_t* aggT  = (bf16_t*)carve((size_t)N_B * HW * C_ALL * 2);

  hipMemsetAsync(KpadT, 0, (size_t)N_B * N_HEAD * PADA * HDIM * 2, stream);
  hipMemsetAsync(Vpad,  0, (size_t)N_B * C_ALL * PADA * 2, stream);

  vb_convert_kernel<<<256, 256, 0, stream>>>(Vb, Vbbf);

  dim3 gp(2, 8, N_B);
  proj_gemm_kernel<<<gp, 256, 0, stream>>>(q, Wq, bq, Qt, 0);
  proj_gemm_kernel<<<gp, 256, 0, stream>>>(k, Wk, bk, KpadT, 1);
  proj_gemm_kernel<<<gp, 256, 0, stream>>>(v, Wv, bv, Vpad, 2);

  rel_gemm_kernel<<<dim3(2, 8, N_B * N_HEAD), 256, 0, stream>>>(Qt, Wrel, brel, relT);
  qk_softmax_kernel<<<dim3(2, 32, N_B * N_HEAD), 64, 0, stream>>>(Qt, KpadT, relT, attnT);
  agg_kernel<<<dim3(2, 32, N_B * N_HEAD), 256, 0, stream>>>(attnT, Vpad, Vbbf, aggT);
  fc_gemm_kernel<<<dim3(2, 32, 1), 256, 0, stream>>>(aggT, Wfc, bfc, out);
}
